// SHA_DIAGONAL_70815420776761
// MI455X (gfx1250) — matrix operand order repaired and checked by reading, not compiled, not run on hardware
//
#include <hip/hip_runtime.h>
#include <hip/hip_bf16.h>

typedef __attribute__((ext_vector_type(16))) __bf16 v16bf;
typedef __attribute__((ext_vector_type(8)))  float  v8f;

#define IN_DIM 4096
#define OUT_DIM 4096
#define NB 4
#define SEQ 2048
#define BS (NB * SEQ)      // 8192 rows
#define R 8
#define RC 4
#define SCALING 4.0f       // ALPHA / R = 32/8

#define BM 128
#define BN 128
#define BK 64
#define LDK (BK + 8)       // bf16 elems per LDS row: 72 -> 144 bytes (16B aligned, 36-bank stride)
#define NKT (IN_DIM / BK)  // 64 K-stages

static __device__ __forceinline__ unsigned short f2bf(float f) {
  unsigned u = __builtin_bit_cast(unsigned, f);
  u += 0x7FFFu + ((u >> 16) & 1u);       // round-to-nearest-even
  return (unsigned short)(u >> 16);
}

union Frag { uint4 q[2]; v16bf v; };

// ---------------- adapter down-projection: down[8192][8] = x @ W_down^T ----
__global__ void __launch_bounds__(256) k_down(const float* __restrict__ x,
                                              const float* __restrict__ Wd,
                                              float* __restrict__ down) {
  const int wave = threadIdx.x >> 5;
  const int lane = threadIdx.x & 31;
  const int row  = blockIdx.x * 8 + wave;          // one wave per row
  const float4* x4 = (const float4*)(x + (size_t)row * IN_DIM);
  const float4* w4 = (const float4*)Wd;            // W_down stays L2/L1 resident (128 KB)
  float acc[R];
#pragma unroll
  for (int r = 0; r < R; ++r) acc[r] = 0.f;
  for (int c = lane; c < IN_DIM / 4; c += 32) {
    float4 xv = x4[c];
#pragma unroll
    for (int r = 0; r < R; ++r) {
      float4 wv = w4[r * (IN_DIM / 4) + c];
      acc[r] += xv.x * wv.x + xv.y * wv.y + xv.z * wv.z + xv.w * wv.w;
    }
  }
#pragma unroll
  for (int r = 0; r < R; ++r)
    for (int off = 16; off > 0; off >>= 1)
      acc[r] += __shfl_xor(acc[r], off, 32);
  if (lane == 0) {
#pragma unroll
    for (int r = 0; r < R; ++r) down[(size_t)row * R + r] = acc[r];
  }
}

// ---------------- adapter attention + W_o: o[8192][8] -----------------------
__global__ void __launch_bounds__(256) k_attn(const float* __restrict__ down,
                                              const float* __restrict__ Wqkv, // [12][8]
                                              const float* __restrict__ Wo,   // [8][4]
                                              float* __restrict__ o) {
  __shared__ float sK[256][RC];
  __shared__ float sV[256][RC];
  __shared__ float sWq[12 * R];
  __shared__ float sWo[R * RC];
  const int b = blockIdx.y;
  const int q = blockIdx.x * 256 + threadIdx.x;    // one thread per query
  if (threadIdx.x < 12 * R) sWq[threadIdx.x] = Wqkv[threadIdx.x];
  if (threadIdx.x < R * RC) sWo[threadIdx.x] = Wo[threadIdx.x];
  __syncthreads();

  float dreg[R];
  {
    const float* dq = down + ((size_t)b * SEQ + q) * R;
#pragma unroll
    for (int r = 0; r < R; ++r) dreg[r] = dq[r];
  }
  float qv[RC];
#pragma unroll
  for (int d = 0; d < RC; ++d) {
    float s = 0.f;
#pragma unroll
    for (int r = 0; r < R; ++r) s += sWq[d * R + r] * dreg[r];
    qv[d] = s;
  }

  float m = -1e30f, l = 0.f;
  float ctx[RC] = {0.f, 0.f, 0.f, 0.f};
  for (int t = 0; t < SEQ / 256; ++t) {
    // each thread materializes one key/value column into LDS
    const float* dk = down + ((size_t)b * SEQ + t * 256 + threadIdx.x) * R;
    float dr[R];
#pragma unroll
    for (int r = 0; r < R; ++r) dr[r] = dk[r];
#pragma unroll
    for (int d = 0; d < RC; ++d) {
      float kk = 0.f, vv = 0.f;
#pragma unroll
      for (int r = 0; r < R; ++r) {
        kk += sWq[(RC + d) * R + r] * dr[r];
        vv += sWq[(2 * RC + d) * R + r] * dr[r];
      }
      sK[threadIdx.x][d] = kk;
      sV[threadIdx.x][d] = vv;
    }
    __syncthreads();
    for (int j = 0; j < 256; ++j) {                // broadcast LDS reads
      float sc = 0.f;
#pragma unroll
      for (int d = 0; d < RC; ++d) sc += qv[d] * sK[j][d];
      sc *= 0.5f;                                  // 1/sqrt(RC)
      float mn = fmaxf(m, sc);
      float corr = __expf(m - mn);
      float p = __expf(sc - mn);
      l = l * corr + p;
#pragma unroll
      for (int d = 0; d < RC; ++d) ctx[d] = ctx[d] * corr + p * sV[j][d];
      m = mn;
    }
    __syncthreads();
  }
  const float inv = 1.f / l;
  float* op = o + ((size_t)b * SEQ + q) * R;
#pragma unroll
  for (int r = 0; r < R; ++r) {
    float s = 0.f;
#pragma unroll
    for (int d = 0; d < RC; ++d) s += sWo[r * RC + d] * ctx[d];
    op[r] = s * inv;
  }
}

// ---------------- main fused GEMM: out = x@Wpre^T + b_pre + diag*scale*(o@Wup^T)
// Double-buffered LDS: global loads for stage kt+1 are issued before the WMMA
// block for stage kt, converted+stored afterwards; one barrier per K-stage.
__global__ void __launch_bounds__(256) k_gemm(const float* __restrict__ x,
                                              const float* __restrict__ Wp,
                                              const float* __restrict__ bpre,
                                              const float* __restrict__ Wup,
                                              const float* __restrict__ diagb,
                                              const float* __restrict__ o,
                                              float* __restrict__ out) {
  __shared__ unsigned short sA[2][BM * LDK];
  __shared__ unsigned short sB[2][BN * LDK];
  const int tid  = threadIdx.x;
  const int wave = tid >> 5;
  const int lane = tid & 31;
  const int wm = wave & 1;        // 2 wave-rows  (64 M each)
  const int wn = wave >> 1;       // 4 wave-cols  (32 N each)
  const int bM = blockIdx.y * BM;
  const int bN = blockIdx.x * BN;

  v8f acc[4][2];
#pragma unroll
  for (int i = 0; i < 4; ++i)
#pragma unroll
    for (int j = 0; j < 2; ++j)
#pragma unroll
      for (int e = 0; e < 8; ++e) acc[i][j][e] = 0.f;

  const int lrow = tid >> 4;   // 0..15
  const int lc4  = tid & 15;   // float4 column within 64-wide K slab

  // ---- prologue: stage tile 0 into buffer 0
#pragma unroll
  for (int j = 0; j < 8; ++j) {
    const int row = lrow + j * 16;
    float4 a = *(const float4*)(x  + (size_t)(bM + row) * IN_DIM + lc4 * 4);
    float4 w = *(const float4*)(Wp + (size_t)(bN + row) * IN_DIM + lc4 * 4);
    unsigned short* pa = &sA[0][row * LDK + lc4 * 4];
    pa[0] = f2bf(a.x); pa[1] = f2bf(a.y); pa[2] = f2bf(a.z); pa[3] = f2bf(a.w);
    unsigned short* pb = &sB[0][row * LDK + lc4 * 4];
    pb[0] = f2bf(w.x); pb[1] = f2bf(w.y); pb[2] = f2bf(w.z); pb[3] = f2bf(w.w);
  }
  __syncthreads();

  int cur = 0;
  for (int kt = 0; kt < NKT; ++kt) {
    const bool more = (kt + 1) < NKT;

    // ---- issue next tile's global loads (in flight during the WMMA block)
    float4 a[8], w[8];
    if (more) {
      const int k0 = (kt + 1) * BK;
#pragma unroll
      for (int j = 0; j < 8; ++j) {
        const int row = lrow + j * 16;
        a[j] = *(const float4*)(x  + (size_t)(bM + row) * IN_DIM + k0 + lc4 * 4);
        w[j] = *(const float4*)(Wp + (size_t)(bN + row) * IN_DIM + k0 + lc4 * 4);
      }
    }
    // ---- L2 prefetch two K-stages ahead (x streams from HBM)
    if (kt + 2 < NKT) {
      const int kp = (kt + 2) * BK;
      const int prow = tid >> 1;                 // 128 rows, 2 lines each
      const int poff = (tid & 1) * 32;           // 128-byte halves of the 256B row slab
      __builtin_prefetch(x  + (size_t)(bM + prow) * IN_DIM + kp + poff, 0, 1);
      __builtin_prefetch(Wp + (size_t)(bN + prow) * IN_DIM + kp + poff, 0, 1);
    }

    // ---- compute on current buffer
#pragma unroll
    for (int ks = 0; ks < 2; ++ks) {              // two K=32 WMMA steps per stage
      const int kw = ks * 32;
      v16bf af[4], bfr[2];
#pragma unroll
      for (int mt = 0; mt < 4; ++mt) {            // A: lane holds row m=lane&15,
        const int row = wm * 64 + mt * 16 + (lane & 15);
        const unsigned short* base = &sA[cur][row * LDK + kw + ((lane >> 4) << 3)];
        Frag u;                                   // K chunks (l/16)*8 and 16+(l/16)*8
        u.q[0] = *(const uint4*)base;
        u.q[1] = *(const uint4*)(base + 16);
        af[mt] = u.v;
      }
#pragma unroll
      for (int nt = 0; nt < 2; ++nt) {            // B: lane holds col n=lane&15,
        const int col = wn * 32 + nt * 16 + (lane & 15);
        const unsigned short* base = &sB[cur][col * LDK + kw + ((lane >> 4) << 3)];
        Frag u;                                   // K chunks (l/16)*8 and 16+(l/16)*8
        u.q[0] = *(const uint4*)base;
        u.q[1] = *(const uint4*)(base + 16);
        bfr[nt] = u.v;
      }
#pragma unroll
      for (int mt = 0; mt < 4; ++mt)
#pragma unroll
        for (int nt = 0; nt < 2; ++nt)
          acc[mt][nt] = __builtin_amdgcn_wmma_f32_16x16x32_bf16(
              false, af[mt], false, bfr[nt], (short)0, acc[mt][nt], false, false);
    }

    // ---- convert + store next tile into the other buffer
    if (more) {
      const int nxt = cur ^ 1;
#pragma unroll
      for (int j = 0; j < 8; ++j) {
        const int row = lrow + j * 16;
        unsigned short* pa = &sA[nxt][row * LDK + lc4 * 4];
        pa[0] = f2bf(a[j].x); pa[1] = f2bf(a[j].y); pa[2] = f2bf(a[j].z); pa[3] = f2bf(a[j].w);
        unsigned short* pb = &sB[nxt][row * LDK + lc4 * 4];
        pb[0] = f2bf(w[j].x); pb[1] = f2bf(w[j].y); pb[2] = f2bf(w[j].z); pb[3] = f2bf(w[j].w);
      }
    }
    __syncthreads();
    cur ^= 1;
  }

  // fused epilogue: + b_pre + SCALING*diag_b*(o . W_up[col])
#pragma unroll
  for (int nt = 0; nt < 2; ++nt) {
    const int col = bN + wn * 32 + nt * 16 + (lane & 15);
    const float4 wu0 = *(const float4*)(Wup + (size_t)col * R);
    const float4 wu1 = *(const float4*)(Wup + (size_t)col * R + 4);
    const float bp = bpre[col];
    const float db = diagb[col] * SCALING;
#pragma unroll
    for (int mt = 0; mt < 4; ++mt) {
#pragma unroll
      for (int r = 0; r < 8; ++r) {
        const int row = bM + wm * 64 + mt * 16 + r + ((lane >> 4) << 3);
        const float4 o0 = *(const float4*)(o + (size_t)row * R);
        const float4 o1 = *(const float4*)(o + (size_t)row * R + 4);
        const float up = o0.x * wu0.x + o0.y * wu0.y + o0.z * wu0.z + o0.w * wu0.w
                       + o1.x * wu1.x + o1.y * wu1.y + o1.z * wu1.z + o1.w * wu1.w;
        out[(size_t)row * OUT_DIM + col] = acc[mt][nt][r] + bp + db * up;
      }
    }
  }
}

extern "C" void kernel_launch(void* const* d_in, const int* in_sizes, int n_in,
                              void* d_out, int out_size, void* d_ws, size_t ws_size,
                              hipStream_t stream) {
  const float* x     = (const float*)d_in[0];
  const float* Wpre  = (const float*)d_in[1];
  const float* bpre  = (const float*)d_in[2];
  const float* Wdown = (const float*)d_in[3];
  const float* Wqkv  = (const float*)d_in[4];
  const float* Wo    = (const float*)d_in[5];
  const float* Wup   = (const float*)d_in[6];
  const float* diagb = (const float*)d_in[7];
  float* out  = (float*)d_out;
  float* down = (float*)d_ws;                     // 8192*8 f32
  float* o    = down + (size_t)BS * R;            // 8192*8 f32

  k_down<<<BS / 8, 256, 0, stream>>>(x, Wdown, down);
  k_attn<<<dim3(SEQ / 256, NB), 256, 0, stream>>>(down, Wqkv, Wo, o);
  k_gemm<<<dim3(OUT_DIM / BN, BS / BM), 256, 0, stream>>>(x, Wpre, bpre, Wup, diagb, o, out);
}
